// SpliceSitesJunctionHead_3539053052316
// MI455X (gfx1250) — matrix operand order repaired and checked by reading, not compiled, not run on hardware
//
#include <hip/hip_runtime.h>
#include <hip/hip_bf16.h>
#include <math.h>

// Problem constants
#define BB 2
#define CC 1536
#define SS 16384
#define HH 768
#define HALF 384
#define TT 367
#define PP 128

typedef __attribute__((ext_vector_type(16))) __bf16 v16bf;
typedef __attribute__((ext_vector_type(8)))  float  v8f;

__device__ __forceinline__ unsigned short f2bf(float f) {
  unsigned int u = __float_as_uint(f);
  unsigned int r = u + 0x7FFFu + ((u >> 16) & 1u);   // round-to-nearest-even
  return (unsigned short)(r >> 16);
}

// A fragment: 16x32 bf16, lane group g (lane>>4), row = tile row + (lane&15)
// halves 0..7  <- K = k0 + 8g .. k0 + 8g+7
// halves 8..15 <- K = k0 + 16 + 8g .. k0 + 16 + 8g+7
__device__ __forceinline__ v16bf load_frag_A(const unsigned short* s, int row, int k0, int ld, int g) {
  union { uint4 u[2]; v16bf v; } un;
  un.u[0] = *(const uint4*)(s + row * ld + k0 + 8 * g);
  un.u[1] = *(const uint4*)(s + row * ld + k0 + 16 + 8 * g);
  return un.v;
}

// B fragment: 32x16 bf16 stored in LDS as [n][k]; lane col = lane&15,
// halves 0..7 <- K = k0 + 8g .. k0 + 8g+7, halves 8..15 <- K = k0 + 16 + 8g .. k0 + 16 + 8g+7
__device__ __forceinline__ v16bf load_frag_B(const unsigned short* s, int col, int k0, int ld, int g) {
  union { uint4 u[2]; v16bf v; } un;
  un.u[0] = *(const uint4*)(s + col * ld + k0 + 8 * g);
  un.u[1] = *(const uint4*)(s + col * ld + k0 + 16 + 8 * g);
  return un.v;
}

// ---------------------------------------------------------------------------
// Stage 1: gathered-column GEMM.
// Glog[b][j][p][h] = sum_c emb[b][c][ssp[b][j][p]] * w[org][h][c] + bias[org][h]
// Grid: 48 WGs = B(2) * mt(4 site types) * nt(6 h-tiles of 128). 256 threads.
// ---------------------------------------------------------------------------
__global__ __launch_bounds__(256) void stage1_gemm(
    const float* __restrict__ emb,     // (B,C,S)
    const float* __restrict__ conv_w,  // (2,H,C)
    const float* __restrict__ conv_b,  // (2,H)
    const int*   __restrict__ orgidx,  // (B)
    const int*   __restrict__ ssp,     // (B,4,P)
    float*       __restrict__ Glog)    // (B,4,P,H)
{
  __shared__ unsigned short sA[128 * 32];
  __shared__ unsigned short sB[128 * 32];

  const int wg = blockIdx.x;
  const int b  = wg / 24;
  const int mt = (wg / 6) % 4;     // site type j (rows mt*128..mt*128+127 of the 512-row gather)
  const int nt = wg % 6;           // h tile
  const int tid  = threadIdx.x;
  const int lane = tid & 31, w = tid >> 5;
  const int g = lane >> 4, ln = lane & 15;
  const int org = orgidx[b];

  v8f acc[8];
  for (int i = 0; i < 8; ++i) acc[i] = (v8f){0.f,0.f,0.f,0.f,0.f,0.f,0.f,0.f};

  for (int kc = 0; kc < CC; kc += 32) {
    __syncthreads();
    // A tile: 128 rows (p) x 32 (c), gathered emb columns, -> bf16
    for (int i = tid; i < 128 * 32; i += 256) {
      int m = i >> 5, k = i & 31;
      int pos = ssp[(b * 4 + mt) * PP + m];
      if (pos < 0) pos = 0;
      float v = emb[((size_t)b * CC + (kc + k)) * SS + pos];
      sA[i] = f2bf(v);
    }
    // B tile stored [n][k]: w[org][nt*128+n][kc+k] -> bf16
    for (int i = tid; i < 128 * 32; i += 256) {
      int n = i >> 5, k = i & 31;
      float v = conv_w[((size_t)org * HH + (nt * 128 + n)) * CC + kc + k];
      sB[i] = f2bf(v);
    }
    __syncthreads();

    v16bf afrag = load_frag_A(sA, w * 16 + ln, 0, 32, g);
    for (int tc = 0; tc < 8; ++tc) {
      v16bf bfrag = load_frag_B(sB, tc * 16 + ln, 0, 32, g);
      acc[tc] = __builtin_amdgcn_wmma_f32_16x16x32_bf16(
          false, afrag, false, bfrag, (short)0, acc[tc], false, false);
    }
  }

  for (int tc = 0; tc < 8; ++tc) {
    for (int r = 0; r < 8; ++r) {
      int m = w * 16 + r + 8 * g;             // p
      int h = nt * 128 + tc * 16 + ln;        // h
      float v = acc[tc][r] + conv_b[org * HH + h];
      Glog[(((size_t)b * 4 + mt) * PP + m) * HH + h] = v;
    }
  }
}

// ---------------------------------------------------------------------------
// Stage 1.5: rope angle tables. ctab/stab[b][j][p][q] for q in [0,384).
// ang = pos * 2^(-20*q/384)   (MAX_POS = 2^20)
// ---------------------------------------------------------------------------
__global__ __launch_bounds__(256) void stage_trig(
    const int* __restrict__ ssp,     // (B,4,P) flattened = B*4*P entries
    float* __restrict__ ctab,
    float* __restrict__ stab)
{
  int idx = blockIdx.x * 256 + threadIdx.x;
  const int N = BB * 4 * PP * HALF;
  if (idx >= N) return;
  int q  = idx % HALF;
  int bp = idx / HALF;
  int pos = ssp[bp];
  if (pos < 0) pos = 0;
  float invf = exp2f(-20.0f * (float)q / (float)HALF);
  float ang  = (float)pos * invf;
  float s, c;
  sincosf(ang, &s, &c);
  ctab[idx] = c;
  stab[idx] = s;
}

// ---------------------------------------------------------------------------
// Stage 2: per (b, t, class) 128x128 GEMM with K=768 over rope-rotated tiles.
// Grid: B*2*T = 1468 WGs, 256 threads (8 waves, wave = d-tile row, 8 a-tiles).
// K processed in 3 chunks of 128 half-pairs (256 h) staged in LDS as bf16.
// Fused epilogue: softplus + junction mask, writes pred and mask outputs.
// ---------------------------------------------------------------------------
__global__ __launch_bounds__(256) void stage2_pairs(
    const float* __restrict__ Glog,   // (B,4,P,H)
    const float* __restrict__ ctab,   // (B,4,P,384)
    const float* __restrict__ stab,
    const float* __restrict__ rope_pos_donor,    // (2,2,T,H)
    const float* __restrict__ rope_pos_acceptor,
    const float* __restrict__ rope_neg_donor,
    const float* __restrict__ rope_neg_acceptor,
    const int*   __restrict__ orgidx,
    const int*   __restrict__ ssp,
    float*       __restrict__ out)    // pred (B,P,P,2T) then mask (B,P,P,2T)
{
  __shared__ unsigned short sD[128 * 256];
  __shared__ unsigned short sA2[128 * 256];

  const int wg  = blockIdx.x;
  const int b   = wg / (2 * TT);
  const int rem = wg % (2 * TT);
  const int cls = rem / TT;          // 0 = pos, 1 = neg
  const int t   = rem % TT;
  const int org = orgidx[b];
  const float* dparam = (cls == 0) ? rope_pos_donor    : rope_neg_donor;
  const float* aparam = (cls == 0) ? rope_pos_acceptor : rope_neg_acceptor;
  const int jd = cls * 2, ja = cls * 2 + 1;

  const int tid  = threadIdx.x;
  const int lane = tid & 31, w = tid >> 5;
  const int g = lane >> 4, ln = lane & 15;

  const size_t scbase = (((size_t)org * 2 + 0) * TT + t) * HH;   // scale row
  const size_t ofbase = (((size_t)org * 2 + 1) * TT + t) * HH;   // offset row

  v8f acc[8];
  for (int i = 0; i < 8; ++i) acc[i] = (v8f){0.f,0.f,0.f,0.f,0.f,0.f,0.f,0.f};

  for (int c = 0; c < 3; ++c) {            // chunks of 128 half-pairs
    __syncthreads();
    const int q0 = c * 128;
    for (int i = tid; i < 2 * 128 * 128; i += 256) {
      int side = i >> 14;                   // 0 = donor(D), 1 = acceptor(A)
      int r    = (i >> 7) & 127;            // p
      int q    = i & 127;                   // local half index
      int j = side ? ja : jd;
      const float* par = side ? aparam : dparam;
      int h1 = q0 + q, h2 = h1 + HALF;
      size_t gbase = (((size_t)b * 4 + j) * PP + r) * HH;
      float x1 = Glog[gbase + h1];
      float x2 = Glog[gbase + h2];
      float u1 = par[scbase + h1] * x1 + par[ofbase + h1];
      float u2 = par[scbase + h2] * x2 + par[ofbase + h2];
      size_t tbase = (((size_t)b * 4 + j) * PP + r) * HALF + h1;
      float cs = ctab[tbase], sn = stab[tbase];
      float r1 = u1 * cs - u2 * sn;          // rotated first-half component
      float r2 = u1 * sn + u2 * cs;          // rotated second-half component
      unsigned short* s = side ? sA2 : sD;
      s[r * 256 + q]       = f2bf(r1);
      s[r * 256 + 128 + q] = f2bf(r2);
    }
    __syncthreads();

    for (int ks = 0; ks < 8; ++ks) {       // 8 k-steps of 32 per chunk
      v16bf afrag = load_frag_A(sD, w * 16 + ln, ks * 32, 256, g);
      for (int tc = 0; tc < 8; ++tc) {
        v16bf bfrag = load_frag_B(sA2, tc * 16 + ln, ks * 32, 256, g);
        acc[tc] = __builtin_amdgcn_wmma_f32_16x16x32_bf16(
            false, afrag, false, bfrag, (short)0, acc[tc], false, false);
      }
    }
  }

  // Epilogue: softplus + mask, fused writes of pred and mask.
  const size_t maskBase = (size_t)BB * PP * PP * (2 * TT);
  for (int tc = 0; tc < 8; ++tc) {
    int a  = tc * 16 + ln;
    int pa = ssp[(b * 4 + ja) * PP + a];
    for (int r = 0; r < 8; ++r) {
      int d  = w * 16 + r + 8 * g;
      int pd = ssp[(b * 4 + jd) * PP + d];
      float x  = acc[tc][r];
      float sp = (x > 20.0f) ? x : log1pf(expf(x));
      bool  mv = (pd >= 0) && (pa >= 0);
      size_t oi = (((size_t)b * PP + d) * PP + a) * (2 * TT) + (size_t)cls * TT + t;
      out[oi]            = mv ? sp : 0.0f;
      out[maskBase + oi] = mv ? 1.0f : 0.0f;
    }
  }
}

extern "C" void kernel_launch(void* const* d_in, const int* in_sizes, int n_in,
                              void* d_out, int out_size, void* d_ws, size_t ws_size,
                              hipStream_t stream) {
  const float* emb    = (const float*)d_in[0];
  const float* conv_w = (const float*)d_in[1];
  const float* conv_b = (const float*)d_in[2];
  const float* rpd    = (const float*)d_in[3];
  const float* rpa    = (const float*)d_in[4];
  const float* rnd    = (const float*)d_in[5];
  const float* rna    = (const float*)d_in[6];
  const int*   orgidx = (const int*)d_in[7];
  const int*   ssp    = (const int*)d_in[8];
  float* out = (float*)d_out;

  float* Glog = (float*)d_ws;                       // B*4*P*H   = 786432 f
  float* ctab = Glog + (size_t)BB * 4 * PP * HH;    // B*4*P*384 = 393216 f
  float* stab = ctab + (size_t)BB * 4 * PP * HALF;  // B*4*P*384 = 393216 f

  stage1_gemm<<<dim3(48), dim3(256), 0, stream>>>(emb, conv_w, conv_b, orgidx, ssp, Glog);

  const int ntrig = BB * 4 * PP * HALF;             // 393216
  stage_trig<<<dim3((ntrig + 255) / 256), dim3(256), 0, stream>>>(ssp, ctab, stab);

  stage2_pairs<<<dim3(BB * 2 * TT), dim3(256), 0, stream>>>(
      Glog, ctab, stab, rpd, rpa, rnd, rna, orgidx, ssp, out);
}
